// FARGANSub_16312285791142
// MI455X (gfx1250) — matrix operand order repaired and checked by reading, not compiled, not run on hardware
//
#include <hip/hip_runtime.h>
#include <hip/hip_bf16.h>

#define BATCH 32768

typedef __attribute__((ext_vector_type(16))) _Float16 v16h;
typedef __attribute__((ext_vector_type(8)))  _Float16 v8h;
typedef __attribute__((ext_vector_type(8)))  float    v8f;

__device__ __forceinline__ float sigm(float x)   { return 1.0f / (1.0f + __expf(-x)); }
__device__ __forceinline__ float tanh_f(float x) { return 1.0f - 2.0f / (__expf(2.0f * x) + 1.0f); }

#define EPI_RAW32  0
#define EPI_TANH16 1
#define EPI_GLU16  2
#define EPI_SIG    3

// Generic WMMA GEMM:  out = epi( A[M x Kpad] * W^T ),  W stored N x Kpad row-major (f16).
// One wave computes one 16x16 tile with v_wmma_f32_16x16x32_f16, f32 accumulate.
template <int EPI>
__global__ __launch_bounds__(256) void gemm_wmma(
    const _Float16* __restrict__ A, int lda,
    const _Float16* __restrict__ W, int ldw,
    int ntiles, int Kpad, int nReal,
    float* __restrict__ o32, int ldo32,
    _Float16* __restrict__ o16, int ldo16,
    const _Float16* __restrict__ aux, int ldaux,
    const float* __restrict__ gain,
    float* __restrict__ o2, int ldo2)
{
    const int lane = threadIdx.x & 31;
    const int wid  = threadIdx.x >> 5;
    const int mtiles = BATCH >> 4;
    const int tile = blockIdx.x * 8 + wid;
    if (tile >= mtiles * ntiles) return;           // wave-uniform: EXEC stays all-1s
    const int mt = tile / ntiles;
    const int nt = tile - mt * ntiles;

    const int hi  = lane >> 4;     // half-wave select
    const int l15 = lane & 15;     // A-row / B-col / C-col

    // A 16x32 f16 layout: lanes 0-15 K=[0..7]+[16..23], lanes 16-31 K=[8..15]+[24..31]
    const _Float16* arow = A + (size_t)(mt * 16 + l15) * lda + hi * 8;
    // B 32x16 layout: lane n holds W[n][k]; lanes 0-15 K=[0..7]+[16..23], lanes 16-31 K=[8..15]+[24..31]
    const _Float16* wrow = W + (size_t)(nt * 16 + l15) * ldw + hi * 8;

    v8f c = {};
    for (int k0 = 0; k0 < Kpad; k0 += 32) {
        v8h a0 = *(const v8h*)(arow + k0);          // global_load_b128
        v8h a1 = *(const v8h*)(arow + k0 + 16);
        v8h b0 = *(const v8h*)(wrow + k0);
        v8h b1 = *(const v8h*)(wrow + k0 + 16);
        v16h av = __builtin_shufflevector(a0, a1, 0,1,2,3,4,5,6,7,8,9,10,11,12,13,14,15);
        v16h bv = __builtin_shufflevector(b0, b1, 0,1,2,3,4,5,6,7,8,9,10,11,12,13,14,15);
        c = __builtin_amdgcn_wmma_f32_16x16x32_f16(false, av, false, bv, (short)0, c, false, false);
    }

    // C/D 16x16 f32: lane -> col l15, VGPR v -> row v + hi*8
    const int row0 = mt * 16 + hi * 8;
    const int col  = nt * 16 + l15;
#pragma unroll
    for (int v = 0; v < 8; ++v) {
        const size_t r = (size_t)(row0 + v);
        float acc = c[v];
        if (EPI == EPI_RAW32) {
            o32[r * ldo32 + col] = acc;
        } else if (EPI == EPI_TANH16) {
            o16[r * ldo16 + col] = (_Float16)tanh_f(acc);
        } else if (EPI == EPI_GLU16) {
            float a = (float)aux[r * ldaux + col];
            o16[r * ldo16 + col] = (_Float16)(a * sigm(acc));
        } else { // EPI_SIG: tanh * gain, dual store (sig_out + exc_mem_new tail)
            if (col < nReal) {
                float vv = tanh_f(acc) * gain[r];
                o32[r * ldo32 + col] = vv;
                o2[r * ldo2 + col]   = vv;
            }
        }
    }
}

// Convert f32 weight(s) -> padded f16, optionally concatenating two matrices along K.
__global__ __launch_bounds__(256) void pack_w(
    const float* __restrict__ Am, int Ka, const float* __restrict__ Bm, int Kb,
    _Float16* __restrict__ dst, int N, int nReal, int Kpad)
{
    int i = blockIdx.x * 256 + threadIdx.x;
    if (i >= N * Kpad) return;
    int n = i / Kpad, k = i - n * Kpad;
    float v = 0.0f;
    if (n < nReal) {
        if (k < Ka)                 v = Am[(size_t)n * Ka + k];
        else if (Bm && k < Ka + Kb) v = Bm[(size_t)n * Kb + (k - Ka)];
    }
    dst[i] = (_Float16)v;
}

// Phase 0: pitch interpolation, gain, tmp/new_s4, xcat (f16), prev/fpitch, output copies.
// One wave32 per row.
__global__ __launch_bounds__(256) void phase0_kernel(
    const float* __restrict__ cond, const float* __restrict__ prev_pred,
    const float* __restrict__ exc,  const float* __restrict__ period,
    const float* __restrict__ s4,   const float* __restrict__ w_gain,
    const float* __restrict__ b_gain,
    float* __restrict__ out_s4, float* __restrict__ out_exc, float* __restrict__ out_pp,
    _Float16* __restrict__ XC, float* __restrict__ PREV,
    float* __restrict__ FPITCH, float* __restrict__ GAIN)
{
    const int row = blockIdx.x * 8 + (threadIdx.x >> 5);
    if (row >= BATCH) return;
    const int lane = threadIdx.x & 31;
    const float* condr = cond + (size_t)row * 80;
    const float* excr  = exc  + (size_t)row * 256;

    float p = period[row];
    if (!(p == p) || fabsf(p) > 1e30f) p = 128.0f;
    p = fminf(fmaxf(p, 32.0f), 255.0f);

    float part = 0.0f;
    for (int j = lane; j < 80; j += 32) part += condr[j] * w_gain[j];
    for (int o = 16; o; o >>= 1) part += __shfl_xor(part, o, 32);
    float g = 0.2f + 0.8f * sigm(part + b_gain[0]);
    if (!(g == g)) g = 1.0f;
    g = fminf(fmaxf(g, 0.001f), 20.0f);
    const float inv = 1.0f / (1e-5f + g);
    if (lane == 0) GAIN[row] = g;

    auto predf = [&](int j) -> float {
        float idxf = 254.0f - p + (float)j;
        float id0  = floorf(idxf);
        float a    = fminf(fmaxf(idxf - id0, 0.0f), 1.0f);
        int i0 = ((int)fminf(fmaxf(id0, 0.0f), 255.0f)) & 255;
        int i1 = ((int)fminf(fmaxf(id0 + 1.0f, 0.0f), 255.0f)) & 255;
        return (1.0f - a) * excr[i0] + a * excr[i1];
    };

    _Float16* xr = XC + (size_t)row * 704;
    float* s4o = out_s4 + (size_t)row * 164;
    for (int k = lane; k < 164; k += 32) xr[k] = (_Float16)s4[(size_t)row * 164 + k];
    for (int k = lane; k < 164; k += 32) {
        float v;
        if (k < 80)       v = condr[k];
        else if (k < 124) v = predf(k - 80) * inv;
        else              v = excr[216 + (k - 124)] * inv;
        s4o[k] = v;                       // new_s4 output
        xr[164 + k] = (_Float16)v;        // xcat = [s4, tmp]
    }
    for (int k = 328 + lane; k < 352; k += 32) xr[k] = (_Float16)0.0f;  // K pad
    for (int j = lane; j < 40; j += 32) {
        PREV[(size_t)row * 40 + j]   = excr[216 + j] * inv;
        FPITCH[(size_t)row * 40 + j] = predf(j + 2) * inv;
    }
    float* excn = out_exc + (size_t)row * 256;
    for (int j = lane; j < 216; j += 32) excn[j] = excr[40 + j];
    float* ppn = out_pp + (size_t)row * 256;
    for (int j = lane; j < 216; j += 32) ppn[j] = prev_pred[(size_t)row * 256 + 40 + j];
    for (int j = lane; j < 40; j += 32) ppn[216 + j] = predf(j + 2);
}

// Build GRU input concat in f16: [main(mc) | pg[g]*fpitch(40) | prev(40) | h(hc) | zeros]
__global__ __launch_bounds__(256) void pack_x(
    const _Float16* __restrict__ mainS, int mc,
    const float* __restrict__ PG, int gate,
    const float* __restrict__ FPITCH, const float* __restrict__ PREV,
    const float* __restrict__ hS, int hc,
    _Float16* __restrict__ XC, int Kpad)
{
    size_t i = (size_t)blockIdx.x * 256 + threadIdx.x;
    if (i >= (size_t)BATCH * Kpad) return;
    int row = (int)(i / Kpad), k = (int)(i - (size_t)row * Kpad);
    _Float16 v;
    if (k < mc)              v = mainS[(size_t)row * mc + k];
    else if (k < mc + 40)    v = (_Float16)(PG[row * 4 + gate] * FPITCH[(size_t)row * 40 + (k - mc)]);
    else if (k < mc + 80)    v = (_Float16)PREV[(size_t)row * 40 + (k - mc - 40)];
    else if (k < mc + 80 + hc) v = (_Float16)hS[(size_t)row * hc + (k - mc - 80)];
    else                     v = (_Float16)0.0f;
    XC[(size_t)row * 704 + k] = v;
}

// skip_in = [g1(160) g2(128) g3(128) fwc0_out(192) pg3*fpitch(40) prev(40) pad(16)]
__global__ __launch_bounds__(256) void pack_skip(
    const _Float16* __restrict__ G1, const _Float16* __restrict__ G2,
    const _Float16* __restrict__ G3, const _Float16* __restrict__ FO,
    const float* __restrict__ PG, const float* __restrict__ FPITCH,
    const float* __restrict__ PREV, _Float16* __restrict__ XC)
{
    size_t i = (size_t)blockIdx.x * 256 + threadIdx.x;
    if (i >= (size_t)BATCH * 704) return;
    int row = (int)(i / 704), k = (int)(i - (size_t)row * 704);
    _Float16 v;
    if (k < 160)      v = G1[(size_t)row * 160 + k];
    else if (k < 288) v = G2[(size_t)row * 128 + (k - 160)];
    else if (k < 416) v = G3[(size_t)row * 128 + (k - 288)];
    else if (k < 608) v = FO[(size_t)row * 192 + (k - 416)];
    else if (k < 648) v = (_Float16)(PG[row * 4 + 3] * FPITCH[(size_t)row * 40 + (k - 608)]);
    else if (k < 688) v = (_Float16)PREV[(size_t)row * 40 + (k - 648)];
    else              v = (_Float16)0.0f;
    XC[i] = v;
}

// pitch_gain = sigmoid(fwc0_out @ w_pg^T + b_pg), tiny N=4 -> one thread per (row,gate)
__global__ __launch_bounds__(256) void pitch_gain_k(
    const _Float16* __restrict__ FO, const float* __restrict__ w_pg,
    const float* __restrict__ b_pg, float* __restrict__ PG)
{
    int i = blockIdx.x * 256 + threadIdx.x;
    if (i >= BATCH * 4) return;
    int row = i >> 2, g = i & 3;
    const _Float16* f = FO + (size_t)row * 192;
    float acc = b_pg[g];
    for (int k = 0; k < 192; ++k) acc += (float)f[k] * w_pg[g * 192 + k];
    PG[i] = sigm(acc);
}

// GRU combine: gsum = [ir+hr | iz+hz | inn+hn], hn separate.  n = tanh(gsum_n - hn + r*hn)
__global__ __launch_bounds__(256) void gru_combine(
    const float* __restrict__ GS, const float* __restrict__ HN,
    const float* __restrict__ hin, int Nh,
    float* __restrict__ hout, _Float16* __restrict__ hout16)
{
    size_t i = (size_t)blockIdx.x * 256 + threadIdx.x;
    if (i >= (size_t)BATCH * Nh) return;
    int row = (int)(i / Nh), j = (int)(i - (size_t)row * Nh);
    const float* gg = GS + (size_t)row * 3 * Nh;
    float r   = sigm(gg[j]);
    float z   = sigm(gg[Nh + j]);
    float hnv = HN[(size_t)row * Nh + j];
    float n   = tanh_f(gg[2 * Nh + j] - hnv + r * hnv);
    float h   = hin[(size_t)row * Nh + j];
    float o   = (1.0f - z) * n + z * h;
    hout[(size_t)row * Nh + j]   = o;
    hout16[(size_t)row * Nh + j] = (_Float16)o;
}

extern "C" void kernel_launch(void* const* d_in, const int* in_sizes, int n_in,
                              void* d_out, int out_size, void* d_ws, size_t ws_size,
                              hipStream_t stream)
{
    (void)in_sizes; (void)n_in; (void)out_size; (void)ws_size;
    const int B = BATCH;
    const float* cond      = (const float*)d_in[0];
    const float* prev_pred = (const float*)d_in[1];
    const float* exc       = (const float*)d_in[2];
    const float* period    = (const float*)d_in[3];
    const float* s1        = (const float*)d_in[4];
    const float* s2        = (const float*)d_in[5];
    const float* s3        = (const float*)d_in[6];
    const float* s4        = (const float*)d_in[7];
    const float* w_gain    = (const float*)d_in[8];
    const float* b_gain    = (const float*)d_in[9];
    const float* w_fwc0    = (const float*)d_in[10];
    const float* w_fwc0g   = (const float*)d_in[11];
    const float* wih1      = (const float*)d_in[12];
    const float* whh1      = (const float*)d_in[13];
    const float* wglu1     = (const float*)d_in[14];
    const float* wih2      = (const float*)d_in[15];
    const float* whh2      = (const float*)d_in[16];
    const float* wglu2     = (const float*)d_in[17];
    const float* wih3      = (const float*)d_in[18];
    const float* whh3      = (const float*)d_in[19];
    const float* wglu3     = (const float*)d_in[20];
    const float* w_skip    = (const float*)d_in[21];
    const float* w_skipg   = (const float*)d_in[22];
    const float* w_sig     = (const float*)d_in[23];
    const float* w_pg      = (const float*)d_in[24];
    const float* b_pg      = (const float*)d_in[25];
    float* out = (float*)d_out;

    // d_out layout (floats): sig(40) exc(256) pp(256) h1(160) h2(128) h3(128) s4(164)
    float* o_sig = out;
    float* o_exc = out + (size_t)B * 40;
    float* o_pp  = out + (size_t)B * 296;
    float* o_h1  = out + (size_t)B * 552;
    float* o_h2  = out + (size_t)B * 712;
    float* o_h3  = out + (size_t)B * 840;
    float* o_s4  = out + (size_t)B * 968;

    size_t off = 0;
    auto take = [&](size_t elems, size_t esz) -> void* {
        void* p = (char*)d_ws + off;
        off += ((elems * esz + 255) / 256) * 256;
        return p;
    };
    _Float16* XC    = (_Float16*)take((size_t)B * 704, 2);   // shared concat buffer
    _Float16* FW16  = (_Float16*)take((size_t)B * 192, 2);
    _Float16* FO16  = (_Float16*)take((size_t)B * 192, 2);
    _Float16* G1    = (_Float16*)take((size_t)B * 160, 2);
    _Float16* G2    = (_Float16*)take((size_t)B * 128, 2);
    _Float16* G3    = (_Float16*)take((size_t)B * 128, 2);
    _Float16* SK16  = (_Float16*)take((size_t)B * 128, 2);
    _Float16* SKG16 = (_Float16*)take((size_t)B * 128, 2);
    _Float16* H116  = (_Float16*)take((size_t)B * 160, 2);
    _Float16* H216  = (_Float16*)take((size_t)B * 128, 2);
    _Float16* H316  = (_Float16*)take((size_t)B * 128, 2);
    float* GS     = (float*)take((size_t)B * 480, 4);
    float* HN     = (float*)take((size_t)B * 160, 4);
    float* PREV   = (float*)take((size_t)B * 40, 4);
    float* FPITCH = (float*)take((size_t)B * 40, 4);
    float* GAIN   = (float*)take((size_t)B, 4);
    float* PG     = (float*)take((size_t)B * 4, 4);
    _Float16* WF0  = (_Float16*)take(192 * 352, 2);
    _Float16* WG0  = (_Float16*)take(192 * 192, 2);
    _Float16* W1   = (_Float16*)take(480 * 448, 2);
    _Float16* WN1  = (_Float16*)take(160 * 160, 2);
    _Float16* WGL1 = (_Float16*)take(160 * 160, 2);
    _Float16* W2   = (_Float16*)take(384 * 384, 2);
    _Float16* WN2  = (_Float16*)take(128 * 128, 2);
    _Float16* WGL2 = (_Float16*)take(128 * 128, 2);
    _Float16* W3   = (_Float16*)take(384 * 352, 2);
    _Float16* WN3  = (_Float16*)take(128 * 128, 2);
    _Float16* WGL3 = (_Float16*)take(128 * 128, 2);
    _Float16* WSK  = (_Float16*)take(128 * 704, 2);
    _Float16* WSKG = (_Float16*)take(128 * 128, 2);
    _Float16* WSIG = (_Float16*)take(48 * 128, 2);

    auto packw = [&](const float* Am, int Ka, const float* Bm, int Kb,
                     _Float16* dst, int N, int nReal, int Kpad) {
        int tot = N * Kpad;
        pack_w<<<(tot + 255) / 256, 256, 0, stream>>>(Am, Ka, Bm, Kb, dst, N, nReal, Kpad);
    };
    auto ggrid = [&](int ntiles) { return dim3((unsigned)(((B >> 4) * ntiles + 7) / 8)); };

    // ---- weight conversion (f32 -> padded f16; GRU gate weights concat [wih|whh]) ----
    packw(w_fwc0, 328, nullptr, 0, WF0, 192, 192, 352);
    packw(w_fwc0g, 192, nullptr, 0, WG0, 192, 192, 192);
    packw(wih1, 272, whh1, 160, W1, 480, 480, 448);
    packw(whh1 + 320 * 160, 160, nullptr, 0, WN1, 160, 160, 160);
    packw(wglu1, 160, nullptr, 0, WGL1, 160, 160, 160);
    packw(wih2, 240, whh2, 128, W2, 384, 384, 384);
    packw(whh2 + 256 * 128, 128, nullptr, 0, WN2, 128, 128, 128);
    packw(wglu2, 128, nullptr, 0, WGL2, 128, 128, 128);
    packw(wih3, 208, whh3, 128, W3, 384, 384, 352);
    packw(whh3 + 256 * 128, 128, nullptr, 0, WN3, 128, 128, 128);
    packw(wglu3, 128, nullptr, 0, WGL3, 128, 128, 128);
    packw(w_skip, 688, nullptr, 0, WSK, 128, 128, 704);
    packw(w_skipg, 128, nullptr, 0, WSKG, 128, 128, 128);
    packw(w_sig, 128, nullptr, 0, WSIG, 48, 40, 128);

    // ---- phase 0 ----
    phase0_kernel<<<B / 8, 256, 0, stream>>>(cond, prev_pred, exc, period, s4, w_gain, b_gain,
                                             o_s4, o_exc, o_pp, XC, PREV, FPITCH, GAIN);

    // fw = tanh(xcat @ w_fwc0^T)   [N=192, K=352]
    gemm_wmma<EPI_TANH16><<<ggrid(12), 256, 0, stream>>>(XC, 704, WF0, 352, 12, 352, 192,
        nullptr, 0, FW16, 192, nullptr, 0, nullptr, nullptr, 0);
    // fwc0_out = fw * sigmoid(fw @ w_fwc0_glu^T)
    gemm_wmma<EPI_GLU16><<<ggrid(12), 256, 0, stream>>>(FW16, 192, WG0, 192, 12, 192, 192,
        nullptr, 0, FO16, 192, FW16, 192, nullptr, nullptr, 0);
    pitch_gain_k<<<(B * 4 + 255) / 256, 256, 0, stream>>>(FO16, w_pg, b_pg, PG);

    // ---- GRU1 (Nh=160) ----
    pack_x<<<(unsigned)(((size_t)B * 448 + 255) / 256), 256, 0, stream>>>(FO16, 192, PG, 0, FPITCH, PREV, s1, 160, XC, 448);
    gemm_wmma<EPI_RAW32><<<ggrid(30), 256, 0, stream>>>(XC, 704, W1, 448, 30, 448, 480,
        GS, 480, nullptr, 0, nullptr, 0, nullptr, nullptr, 0);
    gemm_wmma<EPI_RAW32><<<ggrid(10), 256, 0, stream>>>(XC + 272, 704, WN1, 160, 10, 160, 160,
        HN, 160, nullptr, 0, nullptr, 0, nullptr, nullptr, 0);
    gru_combine<<<(unsigned)(((size_t)B * 160 + 255) / 256), 256, 0, stream>>>(GS, HN, s1, 160, o_h1, H116);
    gemm_wmma<EPI_GLU16><<<ggrid(10), 256, 0, stream>>>(H116, 160, WGL1, 160, 10, 160, 160,
        nullptr, 0, G1, 160, H116, 160, nullptr, nullptr, 0);

    // ---- GRU2 (Nh=128) ----
    pack_x<<<(unsigned)(((size_t)B * 384 + 255) / 256), 256, 0, stream>>>(G1, 160, PG, 1, FPITCH, PREV, s2, 128, XC, 384);
    gemm_wmma<EPI_RAW32><<<ggrid(24), 256, 0, stream>>>(XC, 704, W2, 384, 24, 384, 384,
        GS, 384, nullptr, 0, nullptr, 0, nullptr, nullptr, 0);
    gemm_wmma<EPI_RAW32><<<ggrid(8), 256, 0, stream>>>(XC + 240, 704, WN2, 128, 8, 128, 128,
        HN, 128, nullptr, 0, nullptr, 0, nullptr, nullptr, 0);
    gru_combine<<<(unsigned)(((size_t)B * 128 + 255) / 256), 256, 0, stream>>>(GS, HN, s2, 128, o_h2, H216);
    gemm_wmma<EPI_GLU16><<<ggrid(8), 256, 0, stream>>>(H216, 128, WGL2, 128, 8, 128, 128,
        nullptr, 0, G2, 128, H216, 128, nullptr, nullptr, 0);

    // ---- GRU3 (Nh=128) ----
    pack_x<<<(unsigned)(((size_t)B * 352 + 255) / 256), 256, 0, stream>>>(G2, 128, PG, 2, FPITCH, PREV, s3, 128, XC, 352);
    gemm_wmma<EPI_RAW32><<<ggrid(24), 256, 0, stream>>>(XC, 704, W3, 352, 24, 352, 384,
        GS, 384, nullptr, 0, nullptr, 0, nullptr, nullptr, 0);
    gemm_wmma<EPI_RAW32><<<ggrid(8), 256, 0, stream>>>(XC + 208, 704, WN3, 128, 8, 128, 128,
        HN, 128, nullptr, 0, nullptr, 0, nullptr, nullptr, 0);
    gru_combine<<<(unsigned)(((size_t)B * 128 + 255) / 256), 256, 0, stream>>>(GS, HN, s3, 128, o_h3, H316);
    gemm_wmma<EPI_GLU16><<<ggrid(8), 256, 0, stream>>>(H316, 128, WGL3, 128, 8, 128, 128,
        nullptr, 0, G3, 128, H316, 128, nullptr, nullptr, 0);

    // ---- skip + output ----
    pack_skip<<<(unsigned)(((size_t)B * 704 + 255) / 256), 256, 0, stream>>>(G1, G2, G3, FO16, PG, FPITCH, PREV, XC);
    gemm_wmma<EPI_TANH16><<<ggrid(8), 256, 0, stream>>>(XC, 704, WSK, 704, 8, 704, 128,
        nullptr, 0, SK16, 128, nullptr, 0, nullptr, nullptr, 0);
    gemm_wmma<EPI_GLU16><<<ggrid(8), 256, 0, stream>>>(SK16, 128, WSKG, 128, 8, 128, 128,
        nullptr, 0, SKG16, 128, SK16, 128, nullptr, nullptr, 0);
    // sig_out = tanh(sk @ w_sig^T) * gain  -> o_sig and exc_mem_new[:,216:256]
    gemm_wmma<EPI_SIG><<<ggrid(3), 256, 0, stream>>>(SKG16, 128, WSIG, 128, 3, 128, 40,
        o_sig, 40, nullptr, 0, nullptr, 0, GAIN, o_exc + 216, 256);
}
